// SOABlock_3126736192141
// MI455X (gfx1250) — matrix operand order repaired and checked by reading, not compiled, not run on hardware
//
#include <hip/hip_runtime.h>
#include <hip/hip_bf16.h>
#include <stdint.h>

// ---------------------------------------------------------------------------
// Self-attention block for MI455X (gfx1250), bf16 WMMA pipeline.
//   f = relu(bn(fw@x)); g = relu(bn(gw@x)); h = hw@x + hb
//   attn = softmax(f^T g * mid^-0.5); z = attn @ h^T; out = vw@z + vb + x
// One tiled v_wmma_f32_16x16x32_bf16 GEMM serves all six matmuls.
// A operands: bf16 row-major [M,K] (or f32 for the attn operand).
// B operands: bf16 K-panel-blocked: element (k,n) at ((k>>5)*N + n)*32 + (k&31)
// Staging uses GLOBAL_LOAD_ASYNC_TO_LDS_B128 (ASYNCcnt) when the toolchain
// exposes the builtin; otherwise vectorized b128 copies through VGPRs.
// ---------------------------------------------------------------------------

#if defined(__has_builtin)
#if __has_builtin(__builtin_amdgcn_global_load_async_to_lds_b128)
#define USE_ASYNC_LDS 1
#endif
#endif
#ifndef USE_ASYNC_LDS
#define USE_ASYNC_LDS 0
#endif

typedef __attribute__((ext_vector_type(16))) __bf16 v16bf;
typedef __attribute__((ext_vector_type(8)))  float  v8f;
typedef __attribute__((ext_vector_type(4)))  int    i32x4;

#define BM 128
#define BN 128
#define BK 32
#define KA 40   // padded LDS row stride for A tile (80B rows, 16B aligned)
#define KB 40   // padded LDS row stride for B tile

static __device__ __forceinline__ unsigned short f2bf(float f) {
  union { float f; uint32_t u; } v; v.f = f;
  uint32_t r = v.u + 0x7FFFu + ((v.u >> 16) & 1u);  // round-to-nearest-even
  return (unsigned short)(r >> 16);
}

static __device__ __forceinline__ void wait_async_all() {
#if USE_ASYNC_LDS
#if defined(__has_builtin) && __has_builtin(__builtin_amdgcn_s_wait_asynccnt)
  __builtin_amdgcn_s_wait_asynccnt(0);
#else
  asm volatile("s_wait_asynccnt 0" ::: "memory");
#endif
#endif
}

#if USE_ASYNC_LDS
// Signature (from clang diagnostics): param0 = int4 addrspace(1)*,
// param1 = int4 addrspace(3)*, then i32 offset, i32 cpol.
#define GLD_ASYNC_B128(gsrc, ldst)                                        \
  __builtin_amdgcn_global_load_async_to_lds_b128(                         \
      (__attribute__((address_space(1))) i32x4*)(gsrc),                   \
      (__attribute__((address_space(3))) i32x4*)(ldst), 0, 0)
#endif

union AFrag { v16bf v; uint32_t u[8]; };
union BFrag { v16bf v; uint4 q[2]; };

// storeMode: 0 = f32 row-major O[m*N+n]
//            2 = bf16 transposed row-major O[n*M+m]            (fT)
//            3 = bf16 blocked, k=m, cols N                      (g)
//            4 = bf16 blocked, k=n, cols M                      (hT, zT)
__global__ __launch_bounds__(256) void gemm_bf16_wmma(
    const void* __restrict__ Av, const unsigned short* __restrict__ B,
    void* __restrict__ Ov,
    const float* __restrict__ rowscale, const float* __restrict__ rowbias,
    const float* __restrict__ residual,
    int M, int N, int K, float alpha, int relu, int storeMode,
    int aIsF32,
    long strideA, long strideB, long strideO, long strideR)
{
  const int bz = blockIdx.z;
  const long abase = (long)bz * strideA;
  const long bbase = (long)bz * strideB;
  const long obase = (long)bz * strideO;
  const long rbase = (long)bz * strideR;

  const unsigned short* A16 = (const unsigned short*)Av;
  const float*          A32 = (const float*)Av;
  float*          O32 = (float*)Ov;
  unsigned short* O16 = (unsigned short*)Ov;

  const int m0 = blockIdx.y * BM;
  const int n0 = blockIdx.x * BN;

  __shared__ __align__(16) unsigned short As[2][BM * KA];
  __shared__ __align__(16) unsigned short Bs[2][BN * KB];

  const int tid  = threadIdx.x;
  const int wave = tid >> 5;
  const int lane = tid & 31;
  const int wm   = wave & 1;    // 64-row group within block tile
  const int wn   = wave >> 1;   // 32-col group within block tile
  const int lm   = lane & 15;
  const int hi   = lane >> 4;

  v8f acc[4][2];
#pragma unroll
  for (int mt = 0; mt < 4; ++mt)
#pragma unroll
    for (int nt = 0; nt < 2; ++nt)
      acc[mt][nt] = v8f{};

  // Stage a 128x32 A tile and a 32x128 B tile (blocked layout) into LDS.
  auto stage = [&](int buf, int k0) {
    if (aIsF32) {
#pragma unroll
      for (int i = 0; i < 4; ++i) {
        int slot = tid + 256 * i;            // 1024 slots: 128 m x 8 quads
        int m = slot >> 3, j = (slot & 7) << 2;
        float4 v = *(const float4*)&A32[abase + (long)(m0 + m) * K + (k0 + j)];
        uint2 p;
        p.x = (uint32_t)f2bf(v.x) | ((uint32_t)f2bf(v.y) << 16);
        p.y = (uint32_t)f2bf(v.z) | ((uint32_t)f2bf(v.w) << 16);
        *(uint2*)&As[buf][m * KA + j] = p;
      }
    } else {
#pragma unroll
      for (int i = 0; i < 2; ++i) {
        int slot = tid + 256 * i;            // 512 slots: 128 m x 4 octets
        int m = slot >> 2, j = (slot & 3) << 3;
#if USE_ASYNC_LDS
        GLD_ASYNC_B128(&A16[abase + (long)(m0 + m) * K + (k0 + j)],
                       &As[buf][m * KA + j]);
#else
        *(uint4*)&As[buf][m * KA + j] =
            *(const uint4*)&A16[abase + (long)(m0 + m) * K + (k0 + j)];
#endif
      }
    }
    {
      const long pb = bbase + ((long)(k0 >> 5) * N + n0) * 32;
#pragma unroll
      for (int i = 0; i < 2; ++i) {
        int slot = tid + 256 * i;            // 512 slots: 128 n x 4 octets
        int n = slot >> 2, j = (slot & 3) << 3;
#if USE_ASYNC_LDS
        GLD_ASYNC_B128(&B[pb + (long)n * 32 + j], &Bs[buf][n * KB + j]);
#else
        *(uint4*)&Bs[buf][n * KB + j] = *(const uint4*)&B[pb + (long)n * 32 + j];
#endif
      }
    }
    // Prefetch the K+2 tile into cache (global_prefetch_b8).
    int kp = k0 + BK;
    if (kp < K) {
      if (aIsF32)
        __builtin_prefetch(&A32[abase + (long)(m0 + (tid >> 1)) * K + kp + ((tid & 1) << 4)], 0, 0);
      else
        __builtin_prefetch(&A16[abase + (long)(m0 + (tid >> 1)) * K + kp + ((tid & 1) << 4)], 0, 0);
      __builtin_prefetch(&B[bbase + ((long)(kp >> 5) * N + n0 + (tid >> 1)) * 32 + ((tid & 1) << 4)], 0, 0);
    }
  };

  const int ksteps = K / BK;
  stage(0, 0);
  wait_async_all();
  __syncthreads();

  for (int kt = 0; kt < ksteps; ++kt) {
    const int cur = kt & 1;
    if (kt + 1 < ksteps) stage(cur ^ 1, (kt + 1) * BK);

    // B fragments: lanes 0-15 hold K=0..7,16..23, lanes 16-31 hold K=8..15,24..31 of col n
    BFrag bf[2];
#pragma unroll
    for (int nt = 0; nt < 2; ++nt) {
      int nb  = wn * 32 + nt * 16 + lm;
      int off = nb * KB + 8 * hi;                   // element offset, 16B aligned
      bf[nt].q[0] = *(const uint4*)&Bs[cur][off];
      bf[nt].q[1] = *(const uint4*)&Bs[cur][off + 16];
    }
    // A fragments per ISA 16-bit 16x32 layout:
    // vgpr v: K = (v<4 ? 2v : 16+2(v-4)) + 8*hi, pairs packed per dword
    AFrag af[4];
#pragma unroll
    for (int mt = 0; mt < 4; ++mt) {
      int mb   = wm * 64 + mt * 16 + lm;
      int base = mb * KA + 8 * hi;
#pragma unroll
      for (int v = 0; v < 8; ++v) {
        int kb = (v < 4) ? (2 * v) : (16 + 2 * (v - 4));
        af[mt].u[v] = *(const uint32_t*)&As[cur][base + kb];
      }
    }

#pragma unroll
    for (int mt = 0; mt < 4; ++mt)
#pragma unroll
      for (int nt = 0; nt < 2; ++nt)
        acc[mt][nt] = __builtin_amdgcn_wmma_f32_16x16x32_bf16(
            false, af[mt].v, false, bf[nt].v, (short)0, acc[mt][nt],
            false, false);

    wait_async_all();     // next tile's async copies done before publishing
    __syncthreads();
  }

  // Epilogue. C/D layout: vgpr r -> M = r + 8*hi, N = lane&15.
#pragma unroll
  for (int mt = 0; mt < 4; ++mt) {
#pragma unroll
    for (int nt = 0; nt < 2; ++nt) {
      int gn = n0 + wn * 32 + nt * 16 + lm;
#pragma unroll
      for (int r = 0; r < 8; ++r) {
        int gm = m0 + wm * 64 + mt * 16 + r + 8 * hi;
        float v = acc[mt][nt][r] * alpha;
        if (rowscale)      v = v * rowscale[gm] + rowbias[gm];
        else if (rowbias)  v += rowbias[gm];
        if (relu)          v = fmaxf(v, 0.0f);
        if (residual)      v += residual[rbase + (long)gm * N + gn];
        if (storeMode == 0) {
          O32[obase + (long)gm * N + gn] = v;
        } else if (storeMode == 2) {
          O16[obase + (long)gn * M + gm] = f2bf(v);
        } else if (storeMode == 3) {
          O16[obase + ((long)(gm >> 5) * N + gn) * 32 + (gm & 31)] = f2bf(v);
        } else { // 4
          O16[obase + ((long)(gn >> 5) * M + gm) * 32 + (gn & 31)] = f2bf(v);
        }
      }
    }
  }
}

// Vectorized f32 -> bf16 pack (row-major preserved), for A-side weights.
__global__ __launch_bounds__(256) void pack_bf16(const float* __restrict__ s,
                                                 unsigned short* __restrict__ d,
                                                 long n4) {
  long i = (long)blockIdx.x * 256 + threadIdx.x;
  if (i < n4) {
    float4 v = ((const float4*)s)[i];
    uint2 p;
    p.x = (uint32_t)f2bf(v.x) | ((uint32_t)f2bf(v.y) << 16);
    p.y = (uint32_t)f2bf(v.z) | ((uint32_t)f2bf(v.w) << 16);
    ((uint2*)d)[i] = p;
  }
}

// Pack x [B, C, NPIX] f32 into bf16 K-panel-blocked layout per batch:
// dst[b][(c>>5)*NPIX + n][32] with inner index (c&31).
__global__ __launch_bounds__(256) void pack_x_blocked(const float* __restrict__ x,
                                                      unsigned short* __restrict__ d,
                                                      int C, int NPIX) {
  long t = (long)blockIdx.x * 256 + threadIdx.x;
  int j = (int)(t & 3) << 3;          // k sub-chunk of 8 within 32
  long r = t >> 2;
  int n = (int)(r % NPIX);
  r /= NPIX;
  int cblk = (int)(r % (C >> 5));
  int b = (int)(r / (C >> 5));
  const float* src = x + ((long)b * C + (long)cblk * 32 + j) * NPIX + n;
  unsigned short e[8];
#pragma unroll
  for (int k = 0; k < 8; ++k) e[k] = f2bf(src[(long)k * NPIX]);
  uint4 q;
  q.x = (uint32_t)e[0] | ((uint32_t)e[1] << 16);
  q.y = (uint32_t)e[2] | ((uint32_t)e[3] << 16);
  q.z = (uint32_t)e[4] | ((uint32_t)e[5] << 16);
  q.w = (uint32_t)e[6] | ((uint32_t)e[7] << 16);
  *(uint4*)&d[(long)b * C * NPIX + ((long)cblk * NPIX + n) * 32 + j] = q;
}

// In-place row softmax over rows of length N. One block per row.
__global__ __launch_bounds__(256) void softmax_rows(float* __restrict__ attn, int N) {
  float* p = attn + (long)blockIdx.x * N;
  __shared__ float red[256];
  const int tid = threadIdx.x;

  float mx = -3.0e38f;
  for (int i = tid; i < N; i += 256) mx = fmaxf(mx, p[i]);
  red[tid] = mx; __syncthreads();
  for (int s = 128; s > 0; s >>= 1) {
    if (tid < s) red[tid] = fmaxf(red[tid], red[tid + s]);
    __syncthreads();
  }
  mx = red[0]; __syncthreads();

  float sum = 0.0f;
  for (int i = tid; i < N; i += 256) {
    float e = __expf(p[i] - mx);
    p[i] = e;
    sum += e;
  }
  red[tid] = sum; __syncthreads();
  for (int s = 128; s > 0; s >>= 1) {
    if (tid < s) red[tid] += red[tid + s];
    __syncthreads();
  }
  const float inv = 1.0f / red[0];
  for (int i = tid; i < N; i += 256) p[i] *= inv;
}

// Fold conv bias + BN into per-channel scale/bias.
__global__ void prep_bn(const float* __restrict__ b, const float* __restrict__ gamma,
                        const float* __restrict__ beta, const float* __restrict__ mean,
                        const float* __restrict__ var,
                        float* __restrict__ scale, float* __restrict__ bias, int n) {
  int i = blockIdx.x * blockDim.x + threadIdx.x;
  if (i < n) {
    float inv = gamma[i] * rsqrtf(var[i] + 1e-5f);
    scale[i] = inv;
    bias[i]  = beta[i] + (b[i] - mean[i]) * inv;
  }
}

extern "C" void kernel_launch(void* const* d_in, const int* in_sizes, int n_in,
                              void* d_out, int out_size, void* d_ws, size_t ws_size,
                              hipStream_t stream) {
  (void)in_sizes; (void)n_in; (void)out_size; (void)ws_size;
  const int Bb = 8, C = 2048, H = 48, W = 48;
  const int NPIX = H * W;          // 2304
  const int MID  = C / 4;          // 512

  const float* x       = (const float*)d_in[0];
  const float* f_w     = (const float*)d_in[1];
  const float* f_b     = (const float*)d_in[2];
  const float* f_gamma = (const float*)d_in[3];
  const float* f_beta  = (const float*)d_in[4];
  const float* f_mean  = (const float*)d_in[5];
  const float* f_var   = (const float*)d_in[6];
  const float* g_w     = (const float*)d_in[7];
  const float* g_b     = (const float*)d_in[8];
  const float* g_gamma = (const float*)d_in[9];
  const float* g_beta  = (const float*)d_in[10];
  const float* g_mean  = (const float*)d_in[11];
  const float* g_var   = (const float*)d_in[12];
  const float* h_w     = (const float*)d_in[13];
  const float* h_b     = (const float*)d_in[14];
  const float* v_w     = (const float*)d_in[15];
  const float* v_b     = (const float*)d_in[16];

  float* out  = (float*)d_out;
  float* outz = out;                                  // [B, C, N]
  float* attn = out + (size_t)Bb * C * NPIX;          // [B, N, N]

  // Workspace layout: BN params (f32), then bf16 buffers (16B aligned).
  float* wsf = (float*)d_ws;
  float* fs  = wsf;                 // mid
  float* fbb = fs + MID;
  float* gs  = fbb + MID;
  float* gbb = gs + MID;
  unsigned short* wsu = (unsigned short*)(gbb + MID);

  const size_t xsz  = (size_t)Bb * C * NPIX;   // x elements
  const size_t wsz  = (size_t)MID * C;         // 1x1 conv weight elements
  const size_t per  = (size_t)Bb * NPIX * MID; // per-intermediate elements

  unsigned short* x_bf  = wsu;             wsu += xsz;   // blocked per batch
  unsigned short* fw_bf = wsu;             wsu += wsz;
  unsigned short* gw_bf = wsu;             wsu += wsz;
  unsigned short* hw_bf = wsu;             wsu += wsz;
  unsigned short* vw_bf = wsu;             wsu += wsz;
  unsigned short* fT    = wsu;             wsu += per;   // [B][N, mid] row-major
  unsigned short* gbuf  = wsu;             wsu += per;   // [B] blocked (k=mid, cols N)
  unsigned short* hT    = wsu;             wsu += per;   // [B] blocked (k=pix, cols mid)
  unsigned short* zT    = wsu;             wsu += per;   // [B] blocked (k=mid, cols N)

  const float inv_sqrt_mid = 0.044194173824159216f;   // 512^-0.5

  prep_bn<<<(MID + 255) / 256, 256, 0, stream>>>(f_b, f_gamma, f_beta, f_mean, f_var, fs, fbb, MID);
  prep_bn<<<(MID + 255) / 256, 256, 0, stream>>>(g_b, g_gamma, g_beta, g_mean, g_var, gs, gbb, MID);

  // Pack operands once: x -> blocked bf16, weights -> row-major bf16.
  {
    long nthr = (long)Bb * (C / 32) * NPIX * 4;   // one thread per 8-elem chunk
    pack_x_blocked<<<(unsigned)(nthr / 256), 256, 0, stream>>>(x, x_bf, C, NPIX);
  }
  pack_bf16<<<(unsigned)((wsz / 4 + 255) / 256), 256, 0, stream>>>(f_w, fw_bf, (long)(wsz / 4));
  pack_bf16<<<(unsigned)((wsz / 4 + 255) / 256), 256, 0, stream>>>(g_w, gw_bf, (long)(wsz / 4));
  pack_bf16<<<(unsigned)((wsz / 4 + 255) / 256), 256, 0, stream>>>(h_w, hw_bf, (long)(wsz / 4));
  pack_bf16<<<(unsigned)((wsz / 4 + 255) / 256), 256, 0, stream>>>(v_w, vw_bf, (long)(wsz / 4));

  // 1) f = relu(bn(fw @ x)) -> fT [N, mid] row-major bf16 (A of scores)
  gemm_bf16_wmma<<<dim3(NPIX / BN, MID / BM, Bb), 256, 0, stream>>>(
      fw_bf, x_bf, fT, fs, fbb, nullptr,
      MID, NPIX, C, 1.0f, 1, 2, 0,
      0L, (long)C * NPIX, (long)NPIX * MID, 0L);

  // 2) g = relu(bn(gw @ x)) -> blocked bf16 (B of scores)
  gemm_bf16_wmma<<<dim3(NPIX / BN, MID / BM, Bb), 256, 0, stream>>>(
      gw_bf, x_bf, gbuf, gs, gbb, nullptr,
      MID, NPIX, C, 1.0f, 1, 3, 0,
      0L, (long)C * NPIX, (long)MID * NPIX, 0L);

  // 3) h = hw @ x + hb -> hT blocked bf16 (B of z GEMM, k=pixel)
  gemm_bf16_wmma<<<dim3(NPIX / BN, MID / BM, Bb), 256, 0, stream>>>(
      hw_bf, x_bf, hT, nullptr, h_b, nullptr,
      MID, NPIX, C, 1.0f, 0, 4, 0,
      0L, (long)C * NPIX, (long)NPIX * MID, 0L);

  // 4) scores = fT @ g * mid^-0.5 -> attn region of d_out [N, N] f32
  gemm_bf16_wmma<<<dim3(NPIX / BN, NPIX / BM, Bb), 256, 0, stream>>>(
      fT, gbuf, attn, nullptr, nullptr, nullptr,
      NPIX, NPIX, MID, inv_sqrt_mid, 0, 0, 0,
      (long)NPIX * MID, (long)MID * NPIX, (long)NPIX * NPIX, 0L);

  // 5) softmax rows of attn, in place
  softmax_rows<<<Bb * NPIX, 256, 0, stream>>>(attn, NPIX);

  // 6) z = attn(f32) @ hT -> zT blocked bf16 (B of v conv, k=mid)
  gemm_bf16_wmma<<<dim3(MID / BN, NPIX / BM, Bb), 256, 0, stream>>>(
      attn, hT, zT, nullptr, nullptr, nullptr,
      NPIX, MID, NPIX, 1.0f, 0, 4, 1,
      (long)NPIX * NPIX, (long)NPIX * MID, (long)MID * NPIX, 0L);

  // 7) out = vw @ zT + vb + x -> d_out [C, N] f32
  gemm_bf16_wmma<<<dim3(NPIX / BN, C / BM, Bb), 256, 0, stream>>>(
      vw_bf, zT, outz, nullptr, v_b, x,
      C, NPIX, MID, 1.0f, 0, 0, 0,
      0L, (long)MID * NPIX, (long)C * NPIX, (long)C * NPIX);
}
